// MultiHeadAttention_71390946394869
// MI455X (gfx1250) — matrix operand order repaired and checked by reading, not compiled, not run on hardware
//
#include <hip/hip_runtime.h>
#include <hip/hip_bf16.h>

// ---------------------------------------------------------------------------
// Windowed MHA (Swin-style) fused kernel for gfx1250 (MI455X).
//   B=16, C=128, H=W=128, WINDOW=64 (8x8), heads=4, d=32.
// One workgroup (256 threads = 8 wave32) per (batch, window).
// All GEMMs via v_wmma_f32_16x16x32_bf16; x tile loaded with the Tensor Data
// Mover (tensor_load_to_lds + s_wait_tensorcnt).
// All WMMA operands are stored so each lane's fragment is two contiguous
// 16-byte runs -> ds_load_b128, no strided ds_load_u16 storms.
// ---------------------------------------------------------------------------

typedef __attribute__((ext_vector_type(16))) __bf16        v16bf;
typedef __attribute__((ext_vector_type(8)))  float         v8f;
typedef __attribute__((ext_vector_type(4)))  unsigned int  u32x4;
typedef __attribute__((ext_vector_type(4)))  int           i32x4;
typedef __attribute__((ext_vector_type(8)))  int           i32x8;

#define CCH   128          // channels
#define WIN   64           // tokens per window
#define HD    32           // head dim
#define HWSZ  16384        // H*W

// padded leading dimensions (avoid LDS bank alignment on 64-dword rows)
#define LD128 136          // bf16 row stride for 128-wide matrices
#define LD64V 72           // bf16 row stride for V (64-wide)
#define SLD   66           // f32 row stride for scores
#define PLD   72           // bf16 row stride for probs P
#define OLD   68           // f32 row stride for output tile

// ---- LDS layout (bytes), phase-overlapped; total ~203 KB < 320 KB/WGP -----
#define OFF_WQ  0          // Wq bf16 128 x LD128 (row-major)        34816
#define OFF_WK  34816      // Wk bf16                                 34816
#define OFF_WV  69632      // Wv bf16                                 34816
#define OFF_XF  104448     // x tile f32 128c x 64t (TDM dest)        32768
#define OFF_XT  137216     // x tile bf16 token-major 64 x LD128      17408
#define OFF_Q   154624     // Q^T bf16 64 x LD128 (token-major)       17408
#define OFF_KT  172032     // K^T bf16 64 x LD128 (token-major)       17408
#define OFF_V   189440     // V   bf16 128 x LD64V (channel-major)    18432
#define SMEM_BYTES 207872
// overlays (safe after earlier phases complete):
#define OFF_S   0          // scores f32 4 x 64 x SLD (over Wq/Wk)    67584
#define OFF_P   67584      // P bf16 4 x 64 x PLD (over Wk/Wv tail)   36864
#define OFF_O   104448     // out f32 128 x OLD (over x-f32 + xT)     34816

// ---- wave32 WMMA fragment helpers (A/B use contiguous K: k = 16*half + s) ----------
// A: 16(M) x 32(K). lane m = lane&15. K run: [half*16 .. +15]
__device__ __forceinline__ v16bf load_frag_a(const __bf16* __restrict__ src, int ld) {
    const int lane = threadIdx.x & 31;
    const int m    = lane & 15;
    const int half = lane >> 4;
    const __bf16* p = src + m * ld + half * 16;
    v16bf a;
#pragma unroll
    for (int v = 0; v < 8; ++v) a[v] = p[v];           // one ds_load_b128
#pragma unroll
    for (int v = 0; v < 8; ++v) a[8 + v] = p[8 + v];   // one ds_load_b128
    return a;
}

// B: 32(K) x 16(N) fetched from TRANSPOSED storage BT[n][k] (contiguous K).
// lane n = lane&15; lanes 0-15 hold K=0..15, lanes 16-31 hold K=16..31.
__device__ __forceinline__ v16bf load_frag_bT(const __bf16* __restrict__ srcT, int ld) {
    const int lane = threadIdx.x & 31;
    const int n    = lane & 15;
    const int half = lane >> 4;
    const __bf16* p = srcT + n * ld + half * 16;
    v16bf b;
#pragma unroll
    for (int v = 0; v < 16; ++v) b[v] = p[v];          // two ds_load_b128
    return b;
}

__device__ __forceinline__ void store_acc_bf16(__bf16* __restrict__ dst, int ld,
                                               v8f c, float scale) {
    const int lane = threadIdx.x & 31;
    const int n    = lane & 15;
    const int half = lane >> 4;
#pragma unroll
    for (int r = 0; r < 8; ++r)
        dst[(r + 8 * half) * ld + n] = (__bf16)(c[r] * scale);
}

__device__ __forceinline__ void store_acc_f32(float* __restrict__ dst, int ld, v8f c) {
    const int lane = threadIdx.x & 31;
    const int n    = lane & 15;
    const int half = lane >> 4;
#pragma unroll
    for (int r = 0; r < 8; ++r)
        dst[(r + 8 * half) * ld + n] = c[r];
}

#define WMMA_BF16(A, B, ACC) \
    __builtin_amdgcn_wmma_f32_16x16x32_bf16(false, (A), false, (B), (short)0, (ACC), false, false)

__global__ __launch_bounds__(256)
void swin_mha_fused(const float* __restrict__ x,
                    const float* __restrict__ Wq, const float* __restrict__ bq,
                    const float* __restrict__ Wk, const float* __restrict__ bk,
                    const float* __restrict__ Wv, const float* __restrict__ bv,
                    const float* __restrict__ Bb,
                    float* __restrict__ out) {
    extern __shared__ char smem[];
    const int tid  = threadIdx.x;
    const int wave = tid >> 5;
    const int lane = tid & 31;
    const int n_l  = lane & 15;
    const int half = lane >> 4;

    const int blk = blockIdx.x;
    const int b   = blk >> 8;          // 256 windows per batch
    const int win = blk & 255;
    const int h0  = (win >> 4) * 8;
    const int w0  = (win & 15) * 8;

    __bf16* sWq = (__bf16*)(smem + OFF_WQ);
    __bf16* sWk = (__bf16*)(smem + OFF_WK);
    __bf16* sWv = (__bf16*)(smem + OFF_WV);
    float*  sXf = (float*)(smem + OFF_XF);
    __bf16* sXt = (__bf16*)(smem + OFF_XT);
    __bf16* sQ  = (__bf16*)(smem + OFF_Q);
    __bf16* sKT = (__bf16*)(smem + OFF_KT);
    __bf16* sV  = (__bf16*)(smem + OFF_V);
    float*  sS  = (float*)(smem + OFF_S);
    __bf16* sP  = (__bf16*)(smem + OFF_P);
    float*  sO  = (float*)(smem + OFF_O);

    // ---- Phase 0: TDM async 3-D tile load: x[b, :, h0:h0+8, w0:w0+8] -> LDS f32
    // tile: dim0 = 8 contiguous w, dim1 = 8 h rows (stride 128), dim2 = 128 c
    // (stride 16384). LDS result is channel-major [c][t], t = ty*8+tx.
    if (tid < 32) {                                    // wave 0 issues once
        const float* src = x + (size_t)b * (CCH * HWSZ) + (size_t)h0 * 128 + w0;
        const unsigned long long ga = (unsigned long long)(const void*)src;
        const unsigned lds_addr = (unsigned)(unsigned long long)(void*)sXf;
        u32x4 g0;
        g0.x = 1u;                                     // count=1, user descriptor
        g0.y = lds_addr;                               // lds_addr [63:32]
        g0.z = (unsigned)ga;                           // global_addr[31:0]
        g0.w = (unsigned)((ga >> 32) & 0x1FFFFFFu) | (2u << 30);  // [56:32] | type=2
        i32x8 g1;
        g1.s0 = (2 << 16);                             // data_size=4B, no multicast
        g1.s1 = (128 << 16);                           // tensor_dim0[15:0]=128 (W)
        g1.s2 = (128 << 16);                           // tensor_dim1[15:0]=128 (H)
        g1.s3 = (8 << 16);                             // tile_dim0=8
        g1.s4 = 8 | (128 << 16);                       // tile_dim1=8, tile_dim2=128
        g1.s5 = 128;                                   // tensor_dim0_stride=128
        g1.s6 = (16384 << 16);                         // tensor_dim1_stride=16384
        g1.s7 = 0;
        i32x4 g2 = {128, 0, 0, 0};                     // tensor_dim2=128
        i32x4 g3 = {0, 0, 0, 0};
#if __has_include(<hip/amd_detail/amd_gfx1250_TDM.h>)
        i32x8 g4 = {0, 0, 0, 0, 0, 0, 0, 0};
        __builtin_amdgcn_tensor_load_to_lds(g0, g1, g2, g3, g4, 0);
#else
        __builtin_amdgcn_tensor_load_to_lds(g0, g1, g2, g3, 0);
#endif
    }

    // ---- stage weights (f32 -> bf16, row-major) into LDS while TDM runs ---
    for (int i = tid; i < CCH * CCH; i += 256) {
        const int j = i >> 7, c = i & 127;
        sWq[j * LD128 + c] = (__bf16)Wq[i];
        sWk[j * LD128 + c] = (__bf16)Wk[i];
        sWv[j * LD128 + c] = (__bf16)Wv[i];
    }
    if (tid < 32) __builtin_amdgcn_s_wait_tensorcnt(0);
    __syncthreads();

    // ---- Phase 1: convert x tile to bf16, token-major ---------------------
    for (int i = tid; i < CCH * WIN; i += 256) {
        const int c = i >> 6, t = i & 63;
        sXt[t * LD128 + c] = (__bf16)sXf[c * WIN + t];
    }
    __syncthreads();

    // ---- Phase 2: QKV. Q^T and K^T token-major, V channel-major. ----------
    // 96 16x16 tiles, K=128 (4 WMMAs each); biases in acc init.
    for (int tile = wave; tile < 96; tile += 8) {
        const int which = tile >> 5;                   // 0=Q^T, 1=K^T, 2=V
        const int t     = tile & 31;
        v8f acc;
        if (which < 2) {
            // (Q|K)^T[t, j] = sum_c x[c,t] W[j,c] + b[j]; M=64 tok, N=128 ch
            const __bf16* W   = (which == 0) ? sWq : sWk;
            const float*  bp  = (which == 0) ? bq : bk;
            __bf16*       dst = (which == 0) ? sQ : sKT;
            const float   scl = (which == 0) ? 0.17677669529663687f : 1.0f;
            const int tm = t >> 3, tn = t & 7;
            const float bias = bp[tn * 16 + n_l];      // bias depends on N (=j)
#pragma unroll
            for (int r = 0; r < 8; ++r) acc[r] = bias;
#pragma unroll
            for (int kk = 0; kk < 4; ++kk) {
                v16bf a  = load_frag_a(sXt + (tm * 16) * LD128 + kk * 32, LD128);
                v16bf bf = load_frag_bT(W + (tn * 16) * LD128 + kk * 32, LD128);
                acc = WMMA_BF16(a, bf, acc);
            }
            store_acc_bf16(dst + (tm * 16) * LD128 + tn * 16, LD128, acc, scl);
        } else {
            // V[j, t] = sum_c Wv[j,c] x[c,t] + bv[j]; M=128 ch, N=64 tok
            const int tm = t >> 2, tn = t & 3;
#pragma unroll
            for (int r = 0; r < 8; ++r) acc[r] = bv[tm * 16 + r + 8 * half];
#pragma unroll
            for (int kk = 0; kk < 4; ++kk) {
                v16bf a  = load_frag_a(sWv + (tm * 16) * LD128 + kk * 32, LD128);
                v16bf bf = load_frag_bT(sXt + (tn * 16) * LD128 + kk * 32, LD128);
                acc = WMMA_BF16(a, bf, acc);
            }
            store_acc_bf16(sV + (tm * 16) * LD64V + tn * 16, LD64V, acc, 1.0f);
        }
    }
    __syncthreads();

    // ---- Phase 3: scores = Q^T K (K-depth = d = 32 -> one WMMA per tile) --
    for (int tile = wave; tile < 64; tile += 8) {
        const int h  = tile >> 4;
        const int r  = tile & 15;
        const int tq = r >> 2, tk = r & 3;
        v8f acc;
#pragma unroll
        for (int rr = 0; rr < 8; ++rr)                 // acc init = Bbias[q][k]
            acc[rr] = Bb[(tq * 16 + rr + 8 * half) * WIN + tk * 16 + n_l];
        v16bf a  = load_frag_a(sQ + (tq * 16) * LD128 + h * HD, LD128);
        v16bf bf = load_frag_bT(sKT + (tk * 16) * LD128 + h * HD, LD128);
        acc = WMMA_BF16(a, bf, acc);
        store_acc_f32(sS + h * (WIN * SLD) + (tq * 16) * SLD + tk * 16, SLD, acc);
    }
    __syncthreads();

    // ---- Phase 4: softmax, one (head,row) per thread; write P row-major ---
    {
        const int h = tid >> 6, q = tid & 63;
        float* row = sS + h * (WIN * SLD) + q * SLD;
        float m = -3.4e38f;
        for (int k = 0; k < WIN; ++k) m = fmaxf(m, row[k]);
        float s = 0.f;
        for (int k = 0; k < WIN; ++k) { const float e = __expf(row[k] - m); row[k] = e; s += e; }
        const float inv = 1.f / s;
        __bf16* prow = sP + h * (WIN * PLD) + q * PLD;
        for (int k = 0; k < WIN; ++k) prow[k] = (__bf16)(row[k] * inv);
    }
    __syncthreads();

    // ---- Phase 5: out[d, q] = V P^T ; M=32(d) N=64(q) K=64 (2 WMMAs) ------
    for (int tile = wave; tile < 32; tile += 8) {
        const int h  = tile >> 3;
        const int r  = tile & 7;
        const int tm = r >> 2, tn = r & 3;
        v8f acc;
#pragma unroll
        for (int rr = 0; rr < 8; ++rr) acc[rr] = 0.f;
#pragma unroll
        for (int kk = 0; kk < 2; ++kk) {
            v16bf a  = load_frag_a(sV + (h * HD + tm * 16) * LD64V + kk * 32, LD64V);
            v16bf bf = load_frag_bT(sP + h * (WIN * PLD) + (tn * 16) * PLD + kk * 32, PLD);
            acc = WMMA_BF16(a, bf, acc);
        }
        store_acc_f32(sO + (h * HD + tm * 16) * OLD + tn * 16, OLD, acc);
    }
    __syncthreads();

    // ---- Phase 6: channel-major store back to (B,C,H,W) -------------------
    {
        float* outp = out + (size_t)b * (CCH * HWSZ);
        const int t  = tid & 63;
        const int ty = t >> 3, tx = t & 7;
        for (int i = 0; i < 32; ++i) {
            const int j = i * 4 + (tid >> 6);
            outp[(size_t)j * HWSZ + (h0 + ty) * 128 + (w0 + tx)] = sO[j * OLD + t];
        }
    }
}

extern "C" void kernel_launch(void* const* d_in, const int* in_sizes, int n_in,
                              void* d_out, int out_size, void* d_ws, size_t ws_size,
                              hipStream_t stream) {
    const float* x  = (const float*)d_in[0];
    const float* Wq = (const float*)d_in[1];
    const float* bq = (const float*)d_in[2];
    const float* Wk = (const float*)d_in[3];
    const float* bk = (const float*)d_in[4];
    const float* Wv = (const float*)d_in[5];
    const float* bv = (const float*)d_in[6];
    const float* Bb = (const float*)d_in[7];
    float* out = (float*)d_out;

    (void)in_sizes; (void)n_in; (void)out_size; (void)d_ws; (void)ws_size;

    hipFuncSetAttribute((const void*)swin_mha_fused,
                        hipFuncAttributeMaxDynamicSharedMemorySize, SMEM_BYTES);
    // 16 batches * 256 windows = 4096 workgroups, 256 threads (8 wave32) each
    swin_mha_fused<<<dim3(16 * 256), dim3(256), SMEM_BYTES, stream>>>(
        x, Wq, bq, Wk, bk, Wv, bv, Bb, out);
}
